// AudioLiquidEmberBlock_73134703116726
// MI455X (gfx1250) — matrix operand order repaired and checked by reading, not compiled, not run on hardware
//
#include <hip/hip_runtime.h>
#include <hip/hip_bf16.h>
#include <stdint.h>

// ---------------------------------------------------------------------------
// AudioLiquidEmberBlock for MI455X (gfx1250, wave32, WMMA bf16)
//
//   1) LN1(x) -> normed  (bf16, row-major, A-fragment friendly)
//   2) pack 4 scan weights + W1 + W2 into per-lane WMMA B-fragment layout (bf16)
//   3) 1024x step kernel: 96 waves, each wave = one 16-col tile of UNITS,
//      4 GEMMs (ff1/ff2/ta/tb) sharing one A fragment, fused tanh/sigmoid/LIF
//      epilogue, h written straight back in bf16 fragment-consumable layout.
//      Branch-free inner loop: padded rows 8..15 read a zeroed scratch page.
//   4) LN2 -> h2 (bf16), GEMM1+gelu -> act (bf16), GEMM2 += into d_out
// ---------------------------------------------------------------------------

typedef unsigned short u16;
typedef __attribute__((ext_vector_type(16))) __bf16 v16bf;
typedef __attribute__((ext_vector_type(8)))  float  v8f;

#define N_EMBD   1024
#define UNITS    1536
#define D_IN     2560        // N_EMBD + UNITS
#define TSTEPS   1024
#define BATCH    8
#define NT_SCAN  96          // UNITS/16
#define KB_SCAN  80          // D_IN/32
#define D_FF     4096
#define M_TOK    8192        // 8*1024 tokens

__device__ __forceinline__ u16 f2bf(float f) {
    union { float f; uint32_t u; } v; v.f = f;
    uint32_t u = v.u;
    u += 0x7FFFu + ((u >> 16) & 1u);   // round-to-nearest-even
    return (u16)(u >> 16);
}
__device__ __forceinline__ float sigmoidf_(float x) { return 1.0f / (1.0f + __expf(-x)); }
__device__ __forceinline__ float geluf_(float x) {
    float x3 = x * x * x;
    return 0.5f * x * (1.0f + tanhf(0.7978845608028654f * (x + 0.044715f * x3)));
}

union Frag { v16bf v; uint4 u[2]; };

// A fragment (16x32 bf16): lane L holds row M=L&15; halves 0..7 -> K=koff..koff+7,
// halves 8..15 -> K=koff+16..koff+23, with koff = kb*32 + ((L>=16)?8:0).
__device__ __forceinline__ v16bf load_a_frag(const u16* __restrict__ rowp, int koff) {
    Frag a;
    a.u[0] = *(const uint4*)(rowp + koff);
    a.u[1] = *(const uint4*)(rowp + koff + 16);
    return a.v;
}
// B fragment: pre-packed contiguous 16 halves per lane.
__device__ __forceinline__ v16bf load_b_frag(const u16* __restrict__ p) {
    Frag b;
    b.u[0] = *(const uint4*)(p);
    b.u[1] = *(const uint4*)(p + 8);
    return b.v;
}

// ---------------- LayerNorm (f32 in, bf16 out) -----------------------------
__global__ void ln_bf16_kernel(const float* __restrict__ src, u16* __restrict__ dst,
                               const float* __restrict__ g, const float* __restrict__ b,
                               int cols) {
    const int row = blockIdx.x;
    const float* x = src + (size_t)row * cols;
    __shared__ float s_sum[256], s_sq[256];
    float sum = 0.f, sq = 0.f;
    for (int c = threadIdx.x; c < cols; c += blockDim.x) { float v = x[c]; sum += v; sq += v * v; }
    s_sum[threadIdx.x] = sum; s_sq[threadIdx.x] = sq;
    __syncthreads();
    for (int s = blockDim.x >> 1; s > 0; s >>= 1) {
        if (threadIdx.x < (unsigned)s) { s_sum[threadIdx.x] += s_sum[threadIdx.x + s];
                                         s_sq[threadIdx.x]  += s_sq[threadIdx.x + s]; }
        __syncthreads();
    }
    const float mu   = s_sum[0] / cols;
    const float var  = s_sq[0] / cols - mu * mu;
    const float rstd = rsqrtf(var + 1e-5f);
    for (int c = threadIdx.x; c < cols; c += blockDim.x)
        dst[(size_t)row * cols + c] = f2bf((x[c] - mu) * rstd * g[c] + b[c]);
}

// ---------------- Weight pack: f32 (K,N) -> bf16 B fragments ---------------
__global__ void pack_b_kernel(const float* __restrict__ src, u16* __restrict__ dst,
                              int KB, int N, int total) {
    int idx = blockIdx.x * blockDim.x + threadIdx.x;
    if (idx >= total) return;
    const int h    = idx & 15;
    const int lane = (idx >> 4) & 31;
    const int rest = idx >> 9;
    const int kb   = rest % KB;
    const int nt   = rest / KB;
    const int k = kb * 32 + ((lane >> 4) << 3) + (h & 7) + ((h >> 3) << 4);
    const int n = nt * 16 + (lane & 15);
    dst[idx] = f2bf(src[(size_t)k * N + n]);
}

// ---------------- State init ----------------------------------------------
__global__ void init_state_kernel(const float* __restrict__ hx,
                                  u16* __restrict__ hb0, u16* __restrict__ hb1,
                                  float* __restrict__ hf32, float* __restrict__ memv,
                                  u16* __restrict__ zpage) {
    int id = blockIdx.x * blockDim.x + threadIdx.x;
    if (id < 16 * UNITS) {
        int r = id / UNITS, c = id % UNITS;
        u16 v = (r < BATCH) ? f2bf(hx[r * UNITS + c]) : (u16)0;
        hb0[id] = v; hb1[id] = v;
    }
    if (id < BATCH * UNITS) hf32[id] = hx[id];
    if (id < BATCH * N_EMBD) memv[id] = 0.0f;
    if (id < 32) zpage[id] = (u16)0;          // 64B zero page for padded A rows
}

// ---------------- One recurrent timestep (96 waves) ------------------------
__global__ void __launch_bounds__(128)
step_kernel(int t,
            const u16*  __restrict__ normed,   // [8][1024][1024] bf16
            const u16*  __restrict__ hread,    // [16][1536] bf16 (rows 8..15 = 0)
            u16*        __restrict__ hwrite,   // [16][1536] bf16
            const u16*  __restrict__ Bp,       // [4][96][80][32][16] bf16 fragments
            const u16*  __restrict__ zpage,    // 32 zero halves
            const float* __restrict__ b_ff1, const float* __restrict__ b_ff2,
            const float* __restrict__ b_ta,  const float* __restrict__ b_tb,
            const float* __restrict__ lif_beta, const float* __restrict__ lif_vth,
            const float* __restrict__ x_in,    // [8][1024][1024] f32
            float*      __restrict__ out_x,    // [8][1024][1024] f32 (x + gated)
            float*      __restrict__ memv,     // [8][1024] f32
            float*      __restrict__ hf32)     // [8][1536] f32 (final h)
{
    const int lane  = threadIdx.x & 31;
    const int gwave = blockIdx.x * (blockDim.x >> 5) + (threadIdx.x >> 5);
    if (gwave >= NT_SCAN) return;
    const int nt = gwave;

    const int r         = lane & 15;
    const int koff_lane = (lane >> 4) ? 8 : 0;

    const size_t wstride = (size_t)NT_SCAN * KB_SCAN * 32 * 16;
    const u16* bp0 = Bp + 0 * wstride + (size_t)nt * KB_SCAN * 32 * 16;
    const u16* bp1 = Bp + 1 * wstride + (size_t)nt * KB_SCAN * 32 * 16;
    const u16* bp2 = Bp + 2 * wstride + (size_t)nt * KB_SCAN * 32 * 16;
    const u16* bp3 = Bp + 3 * wstride + (size_t)nt * KB_SCAN * 32 * 16;

    // Branch-free A sources: padded rows 8..15 read the zero page (index scaled by 0)
    const bool rvalid = (r < BATCH);
    const u16* xrow  = rvalid ? (normed + ((size_t)r * TSTEPS + t) * N_EMBD) : zpage;
    const int  kxscl = rvalid ? 1 : 0;
    const u16* hrow  = hread + (size_t)r * UNITS;   // rows 8..15 are real zeros

    v8f acc0 = {}, acc1 = {}, acc2 = {}, acc3 = {};

    // ---- x part of z (K = 0..1023) ----
    for (int kb = 0; kb < 32; ++kb) {
        const int koff = (kb * 32 + koff_lane) * kxscl;
        const v16bf a = load_a_frag(xrow, koff);
        const size_t boff  = ((size_t)kb * 32 + lane) * 16;
        const size_t nboff = ((size_t)(kb + 1) * 32 + lane) * 16;
        __builtin_prefetch((const void*)(bp0 + nboff), 0, 3);   // WGP-scope warm
        __builtin_prefetch((const void*)(bp1 + nboff), 0, 3);
        __builtin_prefetch((const void*)(bp2 + nboff), 0, 3);
        __builtin_prefetch((const void*)(bp3 + nboff), 0, 3);
        acc0 = __builtin_amdgcn_wmma_f32_16x16x32_bf16(false, a, false, load_b_frag(bp0 + boff), (short)0, acc0, false, false);
        acc1 = __builtin_amdgcn_wmma_f32_16x16x32_bf16(false, a, false, load_b_frag(bp1 + boff), (short)0, acc1, false, false);
        acc2 = __builtin_amdgcn_wmma_f32_16x16x32_bf16(false, a, false, load_b_frag(bp2 + boff), (short)0, acc2, false, false);
        acc3 = __builtin_amdgcn_wmma_f32_16x16x32_bf16(false, a, false, load_b_frag(bp3 + boff), (short)0, acc3, false, false);
    }
    // ---- h part of z (K = 1024..2559) ----
    for (int kb = 32; kb < KB_SCAN; ++kb) {
        const int koff = (kb - 32) * 32 + koff_lane;
        const v16bf a = load_a_frag(hrow, koff);
        const size_t boff = ((size_t)kb * 32 + lane) * 16;
        if (kb + 1 < KB_SCAN) {
            const size_t nboff = ((size_t)(kb + 1) * 32 + lane) * 16;
            __builtin_prefetch((const void*)(bp0 + nboff), 0, 3);
            __builtin_prefetch((const void*)(bp1 + nboff), 0, 3);
            __builtin_prefetch((const void*)(bp2 + nboff), 0, 3);
            __builtin_prefetch((const void*)(bp3 + nboff), 0, 3);
        }
        acc0 = __builtin_amdgcn_wmma_f32_16x16x32_bf16(false, a, false, load_b_frag(bp0 + boff), (short)0, acc0, false, false);
        acc1 = __builtin_amdgcn_wmma_f32_16x16x32_bf16(false, a, false, load_b_frag(bp1 + boff), (short)0, acc1, false, false);
        acc2 = __builtin_amdgcn_wmma_f32_16x16x32_bf16(false, a, false, load_b_frag(bp2 + boff), (short)0, acc2, false, false);
        acc3 = __builtin_amdgcn_wmma_f32_16x16x32_bf16(false, a, false, load_b_frag(bp3 + boff), (short)0, acc3, false, false);
    }

    // C/D layout: lanes 0..15 hold rows M=0..7 across the 8 acc elements -> batch rows.
    if (lane < 16) {
        const int col = nt * 16 + lane;
        const float bf1 = b_ff1[col], bf2 = b_ff2[col];
        const float bta = b_ta[col],  btb = b_tb[col];
        const bool docol = (col < N_EMBD);
        const float beta = docol ? sigmoidf_(lif_beta[col]) : 0.0f;
        const float vth  = docol ? lif_vth[col] : 0.0f;
        #pragma unroll
        for (int i = 0; i < BATCH; ++i) {
            const float f1 = tanhf(acc0[i] + bf1);
            const float f2 = tanhf(acc1[i] + bf2);
            const float ti = sigmoidf_((acc2[i] + bta) /* *TS==1 */ + (acc3[i] + btb));
            const float hn = f1 * (1.0f - ti) + ti * f2;
            hwrite[(size_t)i * UNITS + col] = f2bf(hn);
            if (t == TSTEPS - 1) hf32[(size_t)i * UNITS + col] = hn;
            if (docol) {
                const size_t mi = (size_t)i * N_EMBD + col;
                float m = beta * memv[mi] + hn;
                const float sp = sigmoidf_((m - vth) * 5.0f);
                memv[mi] = m - sp * vth;
                const size_t xi = ((size_t)i * TSTEPS + t) * N_EMBD + col;
                out_x[xi] = x_in[xi] + hn * sp;
            }
        }
    }
}

// ---------------- MLP GEMM1: act = gelu(h2 @ W1 + b1), bf16 out ------------
__global__ void __launch_bounds__(128)
gemm1_kernel(const u16* __restrict__ A,       // [8192][1024] bf16
             const u16* __restrict__ Bp,      // W1 fragments, NT=256, KB=32
             const float* __restrict__ bias,  // [4096]
             u16* __restrict__ act)           // [8192][4096] bf16
{
    const int lane  = threadIdx.x & 31;
    const int gwave = blockIdx.x * (blockDim.x >> 5) + (threadIdx.x >> 5);
    const int m_tile  = gwave >> 6;           // 512 row tiles
    const int n_group = gwave & 63;           // 64 groups of 4 col tiles
    if (m_tile >= M_TOK / 16) return;

    const int r         = lane & 15;
    const int koff_lane = (lane >> 4) ? 8 : 0;
    const u16* rowp = A + (size_t)(m_tile * 16 + r) * N_EMBD;

    v8f acc[4] = {{}, {}, {}, {}};
    for (int kb = 0; kb < 32; ++kb) {
        const v16bf a = load_a_frag(rowp, kb * 32 + koff_lane);
        #pragma unroll
        for (int j = 0; j < 4; ++j) {
            const int nt = n_group * 4 + j;
            const u16* bp = Bp + (((size_t)nt * 32 + kb) * 32 + lane) * 16;
            acc[j] = __builtin_amdgcn_wmma_f32_16x16x32_bf16(false, a, false, load_b_frag(bp),
                                                             (short)0, acc[j], false, false);
        }
    }
    const int mofs = (lane >> 4) ? 8 : 0;
    #pragma unroll
    for (int j = 0; j < 4; ++j) {
        const int col = (n_group * 4 + j) * 16 + (lane & 15);
        const float bv = bias[col];
        #pragma unroll
        for (int i = 0; i < 8; ++i) {
            const int row = m_tile * 16 + mofs + i;
            act[(size_t)row * D_FF + col] = f2bf(geluf_(acc[j][i] + bv));
        }
    }
}

// ---------------- MLP GEMM2: out += act @ W2 + b2 --------------------------
__global__ void __launch_bounds__(128)
gemm2_kernel(const u16* __restrict__ A,       // [8192][4096] bf16
             const u16* __restrict__ Bp,      // W2 fragments, NT=64, KB=128
             const float* __restrict__ bias,  // [1024]
             float* __restrict__ out)         // [8192][1024] f32 (+=)
{
    const int lane  = threadIdx.x & 31;
    const int gwave = blockIdx.x * (blockDim.x >> 5) + (threadIdx.x >> 5);
    const int m_tile  = gwave >> 4;           // 512 row tiles
    const int n_group = gwave & 15;           // 16 groups of 4 col tiles
    if (m_tile >= M_TOK / 16) return;

    const int r         = lane & 15;
    const int koff_lane = (lane >> 4) ? 8 : 0;
    const u16* rowp = A + (size_t)(m_tile * 16 + r) * D_FF;

    v8f acc[4] = {{}, {}, {}, {}};
    for (int kb = 0; kb < 128; ++kb) {
        const v16bf a = load_a_frag(rowp, kb * 32 + koff_lane);
        #pragma unroll
        for (int j = 0; j < 4; ++j) {
            const int nt = n_group * 4 + j;
            const u16* bp = Bp + (((size_t)nt * 128 + kb) * 32 + lane) * 16;
            acc[j] = __builtin_amdgcn_wmma_f32_16x16x32_bf16(false, a, false, load_b_frag(bp),
                                                             (short)0, acc[j], false, false);
        }
    }
    const int mofs = (lane >> 4) ? 8 : 0;
    #pragma unroll
    for (int j = 0; j < 4; ++j) {
        const int col = (n_group * 4 + j) * 16 + (lane & 15);
        const float bv = bias[col];
        #pragma unroll
        for (int i = 0; i < 8; ++i) {
            const int row = m_tile * 16 + mofs + i;
            out[(size_t)row * N_EMBD + col] += acc[j][i] + bv;
        }
    }
}

__global__ void copy_h_kernel(const float* __restrict__ hf32, float* __restrict__ dst) {
    int id = blockIdx.x * blockDim.x + threadIdx.x;
    if (id < BATCH * UNITS) dst[id] = hf32[id];
}

// ---------------------------------------------------------------------------
extern "C" void kernel_launch(void* const* d_in, const int* in_sizes, int n_in,
                              void* d_out, int out_size, void* d_ws, size_t ws_size,
                              hipStream_t stream) {
    (void)in_sizes; (void)n_in; (void)out_size; (void)ws_size;
    const float* x        = (const float*)d_in[0];
    const float* hx       = (const float*)d_in[1];
    const float* ln1_g    = (const float*)d_in[2];
    const float* ln1_b    = (const float*)d_in[3];
    const float* W_ff1    = (const float*)d_in[4];
    const float* b_ff1    = (const float*)d_in[5];
    const float* W_ff2    = (const float*)d_in[6];
    const float* b_ff2    = (const float*)d_in[7];
    const float* W_ta     = (const float*)d_in[8];
    const float* b_ta     = (const float*)d_in[9];
    const float* W_tb     = (const float*)d_in[10];
    const float* b_tb     = (const float*)d_in[11];
    const float* lif_beta = (const float*)d_in[12];
    const float* lif_vth  = (const float*)d_in[13];
    const float* ln2_g    = (const float*)d_in[14];
    const float* ln2_b    = (const float*)d_in[15];
    const float* W1       = (const float*)d_in[16];
    const float* b1       = (const float*)d_in[17];
    const float* W2       = (const float*)d_in[18];
    const float* b2       = (const float*)d_in[19];

    float* out   = (float*)d_out;                       // [8][1024][1024] x-part
    float* out_h = out + (size_t)BATCH * TSTEPS * N_EMBD;

    // workspace carve-up (bytes, 256-aligned)
    char* ws = (char*)d_ws;
    size_t off = 0;
    auto take = [&](size_t bytes) { char* p = ws + off; off = (off + bytes + 255) & ~(size_t)255; return p; };
    u16*   normed = (u16*)  take((size_t)BATCH * TSTEPS * N_EMBD * 2);     // 16.8 MB
    u16*   BpScan = (u16*)  take((size_t)4 * D_IN * UNITS * 2);            // 31.5 MB
    u16*   W1p    = (u16*)  take((size_t)N_EMBD * D_FF * 2);               // 8.4 MB
    u16*   W2p    = (u16*)  take((size_t)D_FF * N_EMBD * 2);               // 8.4 MB
    u16*   hb0    = (u16*)  take((size_t)16 * UNITS * 2);
    u16*   hb1    = (u16*)  take((size_t)16 * UNITS * 2);
    u16*   zpage  = (u16*)  take((size_t)32 * 2);
    float* hf32   = (float*)take((size_t)BATCH * UNITS * 4);
    float* memv   = (float*)take((size_t)BATCH * N_EMBD * 4);
    u16*   h2     = (u16*)  take((size_t)M_TOK * N_EMBD * 2);              // 16.8 MB
    u16*   act    = (u16*)  take((size_t)M_TOK * D_FF * 2);                // 67.1 MB

    // 1) LN1 -> bf16
    ln_bf16_kernel<<<M_TOK, 256, 0, stream>>>(x, normed, ln1_g, ln1_b, N_EMBD);

    // 2) weight packs
    {
        const int totS = D_IN * UNITS, gS = (totS + 255) / 256;
        pack_b_kernel<<<gS, 256, 0, stream>>>(W_ff1, BpScan + 0 * (size_t)totS, KB_SCAN, UNITS, totS);
        pack_b_kernel<<<gS, 256, 0, stream>>>(W_ff2, BpScan + 1 * (size_t)totS, KB_SCAN, UNITS, totS);
        pack_b_kernel<<<gS, 256, 0, stream>>>(W_ta,  BpScan + 2 * (size_t)totS, KB_SCAN, UNITS, totS);
        pack_b_kernel<<<gS, 256, 0, stream>>>(W_tb,  BpScan + 3 * (size_t)totS, KB_SCAN, UNITS, totS);
        const int tot1 = N_EMBD * D_FF;
        pack_b_kernel<<<(tot1 + 255) / 256, 256, 0, stream>>>(W1, W1p, 32,  D_FF,   tot1);
        pack_b_kernel<<<(tot1 + 255) / 256, 256, 0, stream>>>(W2, W2p, 128, N_EMBD, tot1);
    }

    // 3) state init (+ zero page)
    init_state_kernel<<<(16 * UNITS + 255) / 256, 256, 0, stream>>>(hx, hb0, hb1, hf32, memv, zpage);

    // 4) sequential scan: one launch per timestep, h ping-pongs
    for (int t = 0; t < TSTEPS; ++t) {
        const u16* hr = (t & 1) ? hb1 : hb0;
        u16*       hw = (t & 1) ? hb0 : hb1;
        step_kernel<<<24, 128, 0, stream>>>(t, normed, hr, hw, BpScan, zpage,
                                            b_ff1, b_ff2, b_ta, b_tb,
                                            lif_beta, lif_vth,
                                            x, out, memv, hf32);
    }

    // 5) LN2 on (x + gated) sitting in d_out
    ln_bf16_kernel<<<M_TOK, 256, 0, stream>>>(out, h2, ln2_g, ln2_b, N_EMBD);

    // 6) MLP
    gemm1_kernel<<<(512 * 64) / 4, 128, 0, stream>>>(h2, W1p, b1, act);
    gemm2_kernel<<<(512 * 16) / 4, 128, 0, stream>>>(act, W2p, b2, out);

    // 7) h_final
    copy_h_kernel<<<(BATCH * UNITS + 255) / 256, 256, 0, stream>>>(hf32, out_h);
}
